// MultiHeadAttention_43989055045766
// MI455X (gfx1250) — matrix operand order repaired and checked by reading, not compiled, not run on hardware
//
#include <hip/hip_runtime.h>

// ---------------- types ----------------
typedef __attribute__((ext_vector_type(16))) __bf16          v16bf;
typedef __attribute__((ext_vector_type(8)))  float           v8f;
typedef __attribute__((ext_vector_type(8)))  float           f8v;
typedef __attribute__((ext_vector_type(8)))  unsigned short  u16x8;
typedef __attribute__((ext_vector_type(16))) unsigned short  u16x16;

union FragU {
    u16x16 u;
    u16x8  h[2];
    v16bf  bf;
};

#define S_LEN   1024
#define BATCH   4
#define DMODEL  1024
#define NHEADS  16
#define DK      64

static __device__ __forceinline__ unsigned short f2bf(float f) {
    union { float f; unsigned int u; } x; x.f = f;
    unsigned int u = x.u;
    u += 0x7FFFu + ((u >> 16) & 1u);          // round-to-nearest-even
    return (unsigned short)(u >> 16);
}

static __device__ __forceinline__ v8f wmma_bf16(const FragU& a, const FragU& b, v8f c) {
    return __builtin_amdgcn_wmma_f32_16x16x32_bf16(false, a.bf, false, b.bf,
                                                   (short)0, c, false, false);
}

// ---------------- fp32 -> bf16 convert ----------------
__global__ void cvt_f32_bf16(const float* __restrict__ src,
                             unsigned short* __restrict__ dst, int n) {
    int i = blockIdx.x * blockDim.x + threadIdx.x;
    int stride = gridDim.x * blockDim.x;
    for (; i < n; i += stride) dst[i] = f2bf(src[i]);
}

// ---------------- projection GEMM: C[4096,1024] = A @ W^T + b ----------------
// one wave -> 64x64 macro tile (4x4 WMMA tiles); 64 x 16 macro tiles = 1024 waves
__global__ void __launch_bounds__(128, 1)
gemm_proj_bf16(const unsigned short* __restrict__ A,
               const unsigned short* __restrict__ W,
               const float* __restrict__ bias,
               float* __restrict__ out_pre,
               float* __restrict__ out_mh,
               unsigned short* __restrict__ out_bf,
               int bf_transposed)
{
    const int lane = threadIdx.x & 31;
    const int wave = threadIdx.x >> 5;
    const int tile = blockIdx.x * 4 + wave;
    const int tm = tile >> 4;                 // 0..63
    const int tn = tile & 15;                 // 0..15
    const int m0 = tm * 64, n0 = tn * 64;

    const int mlane = lane & 15;
    const int k0a = (lane >> 4) << 3;         // A frag: 0 or 8
    const int k0b = (lane >> 4) << 4;         // unused; B frag also uses k0a (0 or 8)

    const unsigned short* arow[4];
    const unsigned short* brow[4];
#pragma unroll
    for (int i = 0; i < 4; ++i) {
        arow[i] = A + (size_t)(m0 + i * 16 + mlane) * DMODEL;
        brow[i] = W + (size_t)(n0 + i * 16 + mlane) * DMODEL;
    }

    v8f acc[4][4];
#pragma unroll
    for (int i = 0; i < 4; ++i)
#pragma unroll
        for (int j = 0; j < 4; ++j) acc[i][j] = (v8f){};

#pragma unroll 2
    for (int kk = 0; kk < DMODEL; kk += 32) {
        FragU a[4], b[4];
#pragma unroll
        for (int i = 0; i < 4; ++i) {
            a[i].h[0] = *(const u16x8*)(arow[i] + kk + k0a);
            a[i].h[1] = *(const u16x8*)(arow[i] + kk + k0a + 16);
        }
#pragma unroll
        for (int j = 0; j < 4; ++j)
        {
            b[j].h[0] = *(const u16x8*)(brow[j] + kk + k0a);
            b[j].h[1] = *(const u16x8*)(brow[j] + kk + k0a + 16);
        }
#pragma unroll
        for (int i = 0; i < 4; ++i)
#pragma unroll
            for (int j = 0; j < 4; ++j)
                acc[i][j] = wmma_bf16(a[i], b[j], acc[i][j]);
    }

    const int mb = (lane >> 4) << 3;
#pragma unroll
    for (int j = 0; j < 4; ++j) {
        const int n  = n0 + j * 16 + (lane & 15);
        const float bn = bias[n];
        const int hh = n >> 6, jj = n & 63;
#pragma unroll
        for (int i = 0; i < 4; ++i) {
#pragma unroll
            for (int r = 0; r < 8; ++r) {
                const int m = m0 + i * 16 + mb + r;   // flat row = s*BATCH + b
                const int s = m >> 2, bidx = m & 3;
                const float v = acc[i][j][r] + bn;
                if (out_pre) out_pre[(size_t)m * DMODEL + n] = v;
                if (out_mh)  out_mh[(((size_t)hh * BATCH + bidx) * S_LEN + s) * DK + jj] = v;
                if (out_bf) {
                    size_t idx = bf_transposed
                        ? ((((size_t)bidx * NHEADS + hh) * DK + jj) * S_LEN + s)
                        : ((((size_t)bidx * NHEADS + hh) * S_LEN + s) * DK + jj);
                    out_bf[idx] = f2bf(v);
                }
            }
        }
    }
}

// ---------------- scores = Q K^T / sqrt(dk), per head ----------------
// one wave -> 64x64 score tile; 64 heads * 16 * 16 = 16384 waves
__global__ void __launch_bounds__(128, 1)
attn_scores(const unsigned short* __restrict__ Qh,
            const unsigned short* __restrict__ Kh,
            float* __restrict__ out_a,
            float* __restrict__ out_b)
{
    const int lane = threadIdx.x & 31;
    const int wave = threadIdx.x >> 5;
    const int tile = blockIdx.x * 4 + wave;
    const int head = tile >> 8;               // b*16 + h
    const int rem  = tile & 255;
    const int s0 = (rem >> 4) * 64, t0 = (rem & 15) * 64;

    const int mlane = lane & 15;
    const int k0a = (lane >> 4) << 3;
    const int k0b = (lane >> 4) << 4;

    const unsigned short* arow[4];
    const unsigned short* brow[4];
#pragma unroll
    for (int i = 0; i < 4; ++i) {
        arow[i] = Qh + ((size_t)head * S_LEN + s0 + i * 16 + mlane) * DK;
        brow[i] = Kh + ((size_t)head * S_LEN + t0 + i * 16 + mlane) * DK;
    }

    v8f acc[4][4];
#pragma unroll
    for (int i = 0; i < 4; ++i)
#pragma unroll
        for (int j = 0; j < 4; ++j) acc[i][j] = (v8f){};

#pragma unroll
    for (int kk = 0; kk < DK; kk += 32) {
        FragU a[4], b[4];
#pragma unroll
        for (int i = 0; i < 4; ++i) {
            a[i].h[0] = *(const u16x8*)(arow[i] + kk + k0a);
            a[i].h[1] = *(const u16x8*)(arow[i] + kk + k0a + 16);
        }
#pragma unroll
        for (int j = 0; j < 4; ++j)
        {
            b[j].h[0] = *(const u16x8*)(brow[j] + kk + k0a);
            b[j].h[1] = *(const u16x8*)(brow[j] + kk + k0a + 16);
        }
#pragma unroll
        for (int i = 0; i < 4; ++i)
#pragma unroll
            for (int j = 0; j < 4; ++j)
                acc[i][j] = wmma_bf16(a[i], b[j], acc[i][j]);
    }

    const int mb = (lane >> 4) << 3;
    const size_t base = (size_t)head * S_LEN * S_LEN;
#pragma unroll
    for (int i = 0; i < 4; ++i) {
#pragma unroll
        for (int r = 0; r < 8; ++r) {
            const int s = s0 + i * 16 + mb + r;
            const size_t rowb = base + (size_t)s * S_LEN;
#pragma unroll
            for (int j = 0; j < 4; ++j) {
                const int n = t0 + j * 16 + (lane & 15);
                const float v = acc[i][j][r] * 0.125f;   // 1/sqrt(64)
                out_a[rowb + n] = v;
                out_b[rowb + n] = v;
            }
        }
    }
}

// ---------------- causal mask tensor ----------------
__global__ void causal_mask_k(float* __restrict__ mask) {
    const int idx = blockIdx.x * blockDim.x + threadIdx.x;   // 1M exact
    const int srow = idx >> 10, t = idx & 1023;
    mask[idx] = (t > srow) ? -__builtin_inff() : 0.0f;
}

// ---------------- masked softmax, one wave per row ----------------
__global__ void softmax_rows(const float* __restrict__ scores,
                             float* __restrict__ out_a,
                             float* __restrict__ out_b)
{
    const int lane = threadIdx.x & 31;
    const int wave = threadIdx.x >> 5;
    const int row  = blockIdx.x * 8 + wave;   // head*1024 + s
    const int s    = row & (S_LEN - 1);
    const size_t base = (size_t)row * S_LEN;

    float v[32];
    float mx = -__builtin_inff();
#pragma unroll
    for (int i = 0; i < 32; ++i) {
        const int col = (i << 5) + lane;
        const float x = scores[base + col];
        v[i] = (col <= s) ? x : -__builtin_inff();
        mx = fmaxf(mx, v[i]);
    }
#pragma unroll
    for (int off = 16; off > 0; off >>= 1)
        mx = fmaxf(mx, __shfl_xor(mx, off, 32));

    float sum = 0.0f;
#pragma unroll
    for (int i = 0; i < 32; ++i) {
        v[i] = __expf(v[i] - mx);             // exp(-inf) -> 0 above diagonal
        sum += v[i];
    }
#pragma unroll
    for (int off = 16; off > 0; off >>= 1)
        sum += __shfl_xor(sum, off, 32);

    const float inv = 1.0f / sum;
#pragma unroll
    for (int i = 0; i < 32; ++i) {
        const int col = (i << 5) + lane;
        const float p = v[i] * inv;
        out_a[base + col] = p;
        out_b[base + col] = p;
    }
}

// ---------------- context = attn @ V, per head; writes concat ----------------
// one wave -> 64 rows x full dk(64); 64 heads * 16 s-tiles = 1024 waves
__global__ void __launch_bounds__(128, 1)
attn_context(const float* __restrict__ attn,
             const unsigned short* __restrict__ VhT,
             float* __restrict__ out_concat,
             unsigned short* __restrict__ out_cbf)
{
    const int lane = threadIdx.x & 31;
    const int wave = threadIdx.x >> 5;
    const int tile = blockIdx.x * 4 + wave;
    const int head = tile >> 4;               // b*16 + h
    const int s0   = (tile & 15) * 64;

    const int mlane = lane & 15;
    const int k0a = (lane >> 4) << 3;
    const int k0b = (lane >> 4) << 4;

    const float* arow[4];
    const unsigned short* brow[4];
#pragma unroll
    for (int i = 0; i < 4; ++i) {
        arow[i] = attn + ((size_t)head * S_LEN + s0 + i * 16 + mlane) * S_LEN;
        brow[i] = VhT + ((size_t)head * DK + i * 16 + mlane) * S_LEN;
    }

    v8f acc[4][4];
#pragma unroll
    for (int i = 0; i < 4; ++i)
#pragma unroll
        for (int j = 0; j < 4; ++j) acc[i][j] = (v8f){};

#pragma unroll 2
    for (int kk = 0; kk < S_LEN; kk += 32) {
        FragU a[4], b[4];
#pragma unroll
        for (int i = 0; i < 4; ++i) {
            const f8v x0 = *(const f8v*)(arow[i] + kk + k0a);
            const f8v x1 = *(const f8v*)(arow[i] + kk + k0a + 16);
#pragma unroll
            for (int e = 0; e < 8; ++e) {
                a[i].u[e]     = f2bf(x0[e]);
                a[i].u[e + 8] = f2bf(x1[e]);
            }
        }
#pragma unroll
        for (int j = 0; j < 4; ++j)
        {
            b[j].h[0] = *(const u16x8*)(brow[j] + kk + k0a);
            b[j].h[1] = *(const u16x8*)(brow[j] + kk + k0a + 16);
        }
#pragma unroll
        for (int i = 0; i < 4; ++i)
#pragma unroll
            for (int j = 0; j < 4; ++j)
                acc[i][j] = wmma_bf16(a[i], b[j], acc[i][j]);
    }

    const int bidx = head >> 4, hh = head & 15;
    const int mb = (lane >> 4) << 3;
#pragma unroll
    for (int i = 0; i < 4; ++i) {
#pragma unroll
        for (int r = 0; r < 8; ++r) {
            const int s = s0 + i * 16 + mb + r;
            const size_t rowb = ((size_t)s * BATCH + bidx) * DMODEL + hh * DK;
#pragma unroll
            for (int j = 0; j < 4; ++j) {
                const int n = j * 16 + (lane & 15);
                const float vv = acc[i][j][r];
                out_concat[rowb + n] = vv;
                out_cbf[rowb + n]    = f2bf(vv);
            }
        }
    }
}

// ---------------- host launcher ----------------
extern "C" void kernel_launch(void* const* d_in, const int* in_sizes, int n_in,
                              void* d_out, int out_size, void* d_ws, size_t ws_size,
                              hipStream_t stream) {
    const float* q  = (const float*)d_in[0];
    const float* k  = (const float*)d_in[1];
    const float* v  = (const float*)d_in[2];
    const float* wq = (const float*)d_in[3];
    const float* bq = (const float*)d_in[4];
    const float* wk = (const float*)d_in[5];
    const float* bk = (const float*)d_in[6];
    const float* wv = (const float*)d_in[7];
    const float* bv = (const float*)d_in[8];
    const float* wo = (const float*)d_in[9];
    const float* bo = (const float*)d_in[10];

    float* out = (float*)d_out;
    // output tuple offsets (floats)
    const size_t SZ1 = (size_t)S_LEN * BATCH * DMODEL;          // 4,194,304
    const size_t SZA = (size_t)BATCH * NHEADS * S_LEN * S_LEN;  // 67,108,864
    const size_t o_output  = 0;
    const size_t o_attn    = o_output  + SZ1;
    const size_t o_Qpre    = o_attn    + SZA;
    const size_t o_Kpre    = o_Qpre    + SZ1;
    const size_t o_Vpre    = o_Kpre    + SZ1;
    const size_t o_scores  = o_Vpre    + SZ1;
    const size_t o_attn2   = o_scores  + SZA;
    const size_t o_Qmh     = o_attn2   + SZA;
    const size_t o_Kmh     = o_Qmh     + SZ1;
    const size_t o_Vmh     = o_Kmh     + SZ1;
    const size_t o_mask    = o_Vmh     + SZ1;
    const size_t o_scores2 = o_mask    + (size_t)S_LEN * S_LEN;
    const size_t o_concat  = o_scores2 + SZA;

    // workspace layout (ushort units)
    unsigned short* ws = (unsigned short*)d_ws;
    unsigned short* qb      = ws;
    unsigned short* kb      = ws + 4194304;
    unsigned short* vb      = ws + 8388608;
    unsigned short* wqb     = ws + 12582912;
    unsigned short* wkb     = ws + 13631488;
    unsigned short* wvb     = ws + 14680064;
    unsigned short* wob     = ws + 15728640;
    unsigned short* Qh      = ws + 16777216;   // [b,h,s,j]
    unsigned short* Kh      = ws + 20971520;   // [b,h,t,j]
    unsigned short* VhT     = ws + 25165824;   // [b,h,j,t]
    unsigned short* concatb = ws + 29360128;   // [s,b,d]

    const dim3 blk256(256);
    const dim3 blk128(128);

    // 1) convert inputs + weights to bf16
    cvt_f32_bf16<<<2048, blk256, 0, stream>>>(q,  qb,  (int)SZ1);
    cvt_f32_bf16<<<2048, blk256, 0, stream>>>(k,  kb,  (int)SZ1);
    cvt_f32_bf16<<<2048, blk256, 0, stream>>>(v,  vb,  (int)SZ1);
    cvt_f32_bf16<<<1024, blk256, 0, stream>>>(wq, wqb, DMODEL * DMODEL);
    cvt_f32_bf16<<<1024, blk256, 0, stream>>>(wk, wkb, DMODEL * DMODEL);
    cvt_f32_bf16<<<1024, blk256, 0, stream>>>(wv, wvb, DMODEL * DMODEL);
    cvt_f32_bf16<<<1024, blk256, 0, stream>>>(wo, wob, DMODEL * DMODEL);

    // 2) Q/K/V projections (1024 macro tiles / 4 waves per block)
    gemm_proj_bf16<<<256, blk128, 0, stream>>>(qb, wqb, bq, out + o_Qpre, out + o_Qmh, Qh, 0);
    gemm_proj_bf16<<<256, blk128, 0, stream>>>(kb, wkb, bk, out + o_Kpre, out + o_Kmh, Kh, 0);
    gemm_proj_bf16<<<256, blk128, 0, stream>>>(vb, wvb, bv, out + o_Vpre, out + o_Vmh, VhT, 1);

    // 3) scores (both pre-mask score outputs are identical)
    attn_scores<<<4096, blk128, 0, stream>>>(Qh, Kh, out + o_scores, out + o_scores2);

    // 4) causal mask tensor + masked softmax
    causal_mask_k<<<4096, blk256, 0, stream>>>(out + o_mask);
    softmax_rows<<<8192, blk256, 0, stream>>>(out + o_scores, out + o_attn, out + o_attn2);

    // 5) context -> concat
    attn_context<<<256, blk128, 0, stream>>>(out + o_attn, VhT, out + o_concat, concatb);

    // 6) output projection
    gemm_proj_bf16<<<256, blk128, 0, stream>>>(concatb, wob, bo, out + o_output,
                                               nullptr, nullptr, 0);
}
